// CausalMultiHeadAttention_30812095381628
// MI455X (gfx1250) — compile-verified, matrix operand order checked by reading, not run on hardware
//
#include <hip/hip_runtime.h>
#include <stdint.h>

// ---------------------------------------------------------------------------
// CDNA5 (gfx1250) causal MHA: bf16 WMMA GEMMs + flash attention.
// D_MODEL=1024, N_HEAD=16, D_K=64, B=2, S=2048.
// ---------------------------------------------------------------------------

#define D_MODEL 1024
#define N_HEAD  16
#define D_K     64
#define BATCH   2
#define SEQ     2048
#define M_TOT   (BATCH * SEQ)      // 4096

typedef __bf16 v16bf __attribute__((ext_vector_type(16)));
typedef float  v8f   __attribute__((ext_vector_type(8)));

union FragAB { uint32_t u[8]; v16bf v; };
union FragC  { float    f[8]; v8f   v; };

__device__ __forceinline__ uint16_t f2bf(float f) {
  union { float f; uint32_t u; } c; c.f = f;
  uint32_t u = c.u;
  uint32_t r = u + 0x7FFFu + ((u >> 16) & 1u);   // round-to-nearest-even
  return (uint16_t)(r >> 16);
}

// uint (=bf16 pair) index within a 16-uint (32 bf16) K-row for the 16-bit
// A/B wave32 WMMA layout: v<4 -> K=2v+8h ; v>=4 -> K=16+2(v-4)+8h
__device__ __forceinline__ int kpat_u(int v, int h) {
  return v + 4 * (h + (v >> 2));
}

// ---------------------------------------------------------------------------
// fp32 -> bf16 conversion
// ---------------------------------------------------------------------------
__global__ void cvt_f32_bf16(const float* __restrict__ src,
                             uint16_t* __restrict__ dst, int n) {
  int i = blockIdx.x * blockDim.x + threadIdx.x;
  if (i < n) dst[i] = f2bf(src[i]);
}

// ---------------------------------------------------------------------------
// Y[M=4096, N=1024] = X(bf16) @ W(bf16)^T + bias
// mode 0: scatter bf16 result to [b, head, s, dk]   (QKV projections)
// mode 1: fp32 result row-major to outf             (output projection)
// Tile: BM=128, BN=128, BK=32; 8 waves; each wave: 2x4 16x16 subtiles.
// ---------------------------------------------------------------------------
#define BM 128
#define BN 128
#define BK 32

__global__ __launch_bounds__(256) void gemm_bf16_wmma(
    const uint16_t* __restrict__ X, const uint16_t* __restrict__ W,
    const float* __restrict__ bias, uint16_t* __restrict__ outb,
    float* __restrict__ outf, int mode) {
  __shared__ uint32_t ldsX[BM][BK / 2];   // 128 x 16 uints (32 bf16/row)
  __shared__ uint32_t ldsW[BN][BK / 2];

  const int m0   = blockIdx.y * BM;
  const int n0   = blockIdx.x * BN;
  const int tid  = threadIdx.x;
  const int wave = tid >> 5;
  const int lane = tid & 31;
  const int wm   = wave >> 1;   // 0..3  -> 32-row strip
  const int wn   = wave & 1;    // 0..1  -> 64-col strip
  const int h2   = lane >> 4;
  const int l16  = lane & 15;

  FragC acc[2][4];
#pragma unroll
  for (int mi = 0; mi < 2; ++mi)
#pragma unroll
    for (int ni = 0; ni < 4; ++ni)
#pragma unroll
      for (int r = 0; r < 8; ++r) acc[mi][ni].f[r] = 0.0f;

  for (int kb = 0; kb < D_MODEL; kb += BK) {
    // cooperative tile load: each of 256 threads moves 32B of X and 32B of W
    {
      const int row = tid >> 1;        // 0..127
      const int seg = tid & 1;         // 0..1 (8 uints each)
      const uint4* gx =
          (const uint4*)(X + (size_t)(m0 + row) * D_MODEL + kb + seg * 16);
      uint4 a0 = gx[0], a1 = gx[1];
      *(uint4*)&ldsX[row][seg * 8 + 0] = a0;
      *(uint4*)&ldsX[row][seg * 8 + 4] = a1;
      const uint4* gw =
          (const uint4*)(W + (size_t)(n0 + row) * D_MODEL + kb + seg * 16);
      uint4 b0 = gw[0], b1 = gw[1];
      *(uint4*)&ldsW[row][seg * 8 + 0] = b0;
      *(uint4*)&ldsW[row][seg * 8 + 4] = b1;
    }
    __syncthreads();

    FragAB aF[2], bF[4];
#pragma unroll
    for (int mi = 0; mi < 2; ++mi) {
      const int row = wm * 32 + mi * 16 + l16;
#pragma unroll
      for (int v = 0; v < 8; ++v) aF[mi].u[v] = ldsX[row][kpat_u(v, h2)];
    }
#pragma unroll
    for (int ni = 0; ni < 4; ++ni) {
      const int row = wn * 64 + ni * 16 + l16;
#pragma unroll
      for (int v = 0; v < 8; ++v) bF[ni].u[v] = ldsW[row][kpat_u(v, h2)];
    }
#pragma unroll
    for (int mi = 0; mi < 2; ++mi)
#pragma unroll
      for (int ni = 0; ni < 4; ++ni)
        acc[mi][ni].v = __builtin_amdgcn_wmma_f32_16x16x32_bf16(
            false, aF[mi].v, false, bF[ni].v, (short)0, acc[mi][ni].v, false,
            false);
    __syncthreads();
  }

  // epilogue: C layout -> lanes 0-15: M=r, lanes 16-31: M=8+r, N=lane%16
#pragma unroll
  for (int mi = 0; mi < 2; ++mi)
#pragma unroll
    for (int ni = 0; ni < 4; ++ni)
#pragma unroll
      for (int r = 0; r < 8; ++r) {
        const int gm = m0 + wm * 32 + mi * 16 + h2 * 8 + r;
        const int gn = n0 + wn * 64 + ni * 16 + l16;
        const float val = acc[mi][ni].f[r] + bias[gn];
        if (mode == 0) {
          const int b  = gm >> 11;       // / SEQ
          const int s  = gm & 2047;
          const int hh = gn >> 6;        // / D_K
          const int dk = gn & 63;
          outb[(((size_t)(b * N_HEAD + hh) * SEQ + s) << 6) + dk] = f2bf(val);
        } else {
          outf[(size_t)gm * D_MODEL + gn] = val;
        }
      }
}

// ---------------------------------------------------------------------------
// Flash attention, causal. Block = (b*H, 128-row q block), 8 waves.
// Each wave: 16 q rows. K/V staged in LDS per 32-key step (V transposed).
// ---------------------------------------------------------------------------
__global__ __launch_bounds__(256) void attn_flash_wmma(
    const uint16_t* __restrict__ Qb, const uint16_t* __restrict__ Kb,
    const uint16_t* __restrict__ Vb, uint16_t* __restrict__ ctxb) {
  __shared__ uint32_t ldsK[32][32];        // 32 keys x 64 bf16
  __shared__ uint32_t ldsVT[64][16];       // transposed: [dk][key] 64 x 32 bf16
  __shared__ uint32_t ldsP[8][16][16];     // per-wave P tile 16 x 32 bf16

  const int bh   = blockIdx.y;             // 0..31  (b*16 + head)
  const int q0   = blockIdx.x * 128;
  const int tid  = threadIdx.x;
  const int wave = tid >> 5;
  const int lane = tid & 31;
  const int h2   = lane >> 4;
  const int l16  = lane & 15;
  const size_t base = (size_t)bh * SEQ * D_K;

  // Q A-fragments (16 rows x 64 dk -> two 16x32 fragments), loaded once
  FragAB aQ[2];
  {
    const int qrow = q0 + wave * 16 + l16;
    const uint32_t* qp = (const uint32_t*)(Qb + base + (size_t)qrow * D_K);
#pragma unroll
    for (int f = 0; f < 2; ++f)
#pragma unroll
      for (int v = 0; v < 8; ++v) aQ[f].u[v] = qp[f * 16 + kpat_u(v, h2)];
  }

  FragC o[4];
#pragma unroll
  for (int ni = 0; ni < 4; ++ni)
#pragma unroll
    for (int r = 0; r < 8; ++r) o[ni].f[r] = 0.0f;
  float mrun[8], lrun[8];
#pragma unroll
  for (int r = 0; r < 8; ++r) { mrun[r] = -1e30f; lrun[r] = 0.0f; }

  const int kend = q0 + 128;               // causal: keys <= max q row
  for (int kb = 0; kb < kend; kb += 32) {
    // cooperative K/V tile load (each thread: 16B of K, 16B of V)
    {
      const int row = tid >> 3;            // 0..31
      const int seg = tid & 7;             // 8 bf16 = 4 uints
      const uint4 kk =
          *(const uint4*)(Kb + base + (size_t)(kb + row) * D_K + seg * 8);
      *(uint4*)&ldsK[row][seg * 4] = kk;
      const uint4 vv =
          *(const uint4*)(Vb + base + (size_t)(kb + row) * D_K + seg * 8);
      uint32_t tmp[4] = {vv.x, vv.y, vv.z, vv.w};
      uint16_t* vt16 = (uint16_t*)ldsVT;
#pragma unroll
      for (int j = 0; j < 4; ++j) {
        const int dk0 = seg * 8 + j * 2;
        vt16[(dk0 + 0) * 32 + row] = (uint16_t)(tmp[j] & 0xFFFFu);
        vt16[(dk0 + 1) * 32 + row] = (uint16_t)(tmp[j] >> 16);
      }
      if (kb + 32 < kend) {                // prefetch next tile
        __builtin_prefetch(
            (const void*)(Kb + base + (size_t)(kb + 32 + row) * D_K), 0, 0);
        __builtin_prefetch(
            (const void*)(Vb + base + (size_t)(kb + 32 + row) * D_K), 0, 0);
      }
    }
    __syncthreads();

    // scores S = Q @ K^T (16 x 32), two 16x16 subtiles
    FragAB bK[2][2];
#pragma unroll
    for (int nj = 0; nj < 2; ++nj)
#pragma unroll
      for (int f = 0; f < 2; ++f)
#pragma unroll
        for (int v = 0; v < 8; ++v)
          bK[nj][f].u[v] = ldsK[nj * 16 + l16][f * 16 + kpat_u(v, h2)];

    FragC s[2];
#pragma unroll
    for (int nj = 0; nj < 2; ++nj) {
#pragma unroll
      for (int r = 0; r < 8; ++r) s[nj].f[r] = 0.0f;
      s[nj].v = __builtin_amdgcn_wmma_f32_16x16x32_bf16(
          false, aQ[0].v, false, bK[nj][0].v, (short)0, s[nj].v, false, false);
      s[nj].v = __builtin_amdgcn_wmma_f32_16x16x32_bf16(
          false, aQ[1].v, false, bK[nj][1].v, (short)0, s[nj].v, false, false);
    }

    // scale + causal mask
#pragma unroll
    for (int nj = 0; nj < 2; ++nj)
#pragma unroll
      for (int r = 0; r < 8; ++r) {
        const int col = kb + nj * 16 + l16;
        const int row = q0 + wave * 16 + h2 * 8 + r;
        float sv = s[nj].f[r] * 0.125f;    // 1/sqrt(64)
        if (col > row) sv = -1e30f;
        s[nj].f[r] = sv;
      }

    // online softmax (reductions stay within 16-lane halves)
    float mnew[8], rs[8];
#pragma unroll
    for (int r = 0; r < 8; ++r) {
      float mx = fmaxf(s[0].f[r], s[1].f[r]);
      mx = fmaxf(mx, __shfl_xor(mx, 1, 16));
      mx = fmaxf(mx, __shfl_xor(mx, 2, 16));
      mx = fmaxf(mx, __shfl_xor(mx, 4, 16));
      mx = fmaxf(mx, __shfl_xor(mx, 8, 16));
      mnew[r] = fmaxf(mrun[r], mx);
    }
#pragma unroll
    for (int nj = 0; nj < 2; ++nj)
#pragma unroll
      for (int r = 0; r < 8; ++r)
        s[nj].f[r] = __expf(s[nj].f[r] - mnew[r]);
#pragma unroll
    for (int r = 0; r < 8; ++r) {
      float sum = s[0].f[r] + s[1].f[r];
      sum += __shfl_xor(sum, 1, 16);
      sum += __shfl_xor(sum, 2, 16);
      sum += __shfl_xor(sum, 4, 16);
      sum += __shfl_xor(sum, 8, 16);
      rs[r] = sum;
      const float cf = __expf(mrun[r] - mnew[r]);
      lrun[r] = lrun[r] * cf + rs[r];
      mrun[r] = mnew[r];
#pragma unroll
      for (int ni = 0; ni < 4; ++ni) o[ni].f[r] *= cf;
    }

    // P: C-layout -> per-wave LDS -> A-layout reload
    {
      uint16_t* p16 = (uint16_t*)ldsP[wave];
#pragma unroll
      for (int nj = 0; nj < 2; ++nj)
#pragma unroll
        for (int r = 0; r < 8; ++r)
          p16[(h2 * 8 + r) * 32 + nj * 16 + l16] = f2bf(s[nj].f[r]);
    }
    asm volatile("s_wait_dscnt 0" ::: "memory");

    FragAB aP;
#pragma unroll
    for (int v = 0; v < 8; ++v) aP.u[v] = ldsP[wave][l16][kpat_u(v, h2)];

    FragAB bV[4];
#pragma unroll
    for (int ni = 0; ni < 4; ++ni)
#pragma unroll
      for (int v = 0; v < 8; ++v)
        bV[ni].u[v] = ldsVT[ni * 16 + l16][kpat_u(v, h2)];

#pragma unroll
    for (int ni = 0; ni < 4; ++ni)
      o[ni].v = __builtin_amdgcn_wmma_f32_16x16x32_bf16(
          false, aP.v, false, bV[ni].v, (short)0, o[ni].v, false, false);

    __syncthreads();
  }

  // normalize + write ctx in [b, s, head*64+dk] bf16 layout
  const int b  = bh >> 4;
  const int hh = bh & 15;
#pragma unroll
  for (int ni = 0; ni < 4; ++ni)
#pragma unroll
    for (int r = 0; r < 8; ++r) {
      const int row = q0 + wave * 16 + h2 * 8 + r;
      const float val = o[ni].f[r] / lrun[r];
      const size_t m = (size_t)b * SEQ + row;
      const int col = hh * 64 + ni * 16 + l16;
      ctxb[m * D_MODEL + col] = f2bf(val);
    }
}

// ---------------------------------------------------------------------------
// Launcher
// ---------------------------------------------------------------------------
extern "C" void kernel_launch(void* const* d_in, const int* in_sizes, int n_in,
                              void* d_out, int out_size, void* d_ws,
                              size_t ws_size, hipStream_t stream) {
  (void)in_sizes; (void)n_in; (void)out_size; (void)ws_size;
  const float* x  = (const float*)d_in[0];
  const float* Wq = (const float*)d_in[1];
  const float* bq = (const float*)d_in[2];
  const float* Wk = (const float*)d_in[3];
  const float* bk = (const float*)d_in[4];
  const float* Wv = (const float*)d_in[5];
  const float* bv = (const float*)d_in[6];
  const float* Wo = (const float*)d_in[7];
  const float* bo = (const float*)d_in[8];

  uint8_t* ws = (uint8_t*)d_ws;
  constexpr size_t X_ELEMS = (size_t)M_TOT * D_MODEL;      // 4096*1024
  constexpr size_t W_ELEMS = (size_t)D_MODEL * D_MODEL;    // 1024*1024
  constexpr size_t QKV_ELEMS = (size_t)BATCH * N_HEAD * SEQ * D_K;

  uint16_t* xb   = (uint16_t*)(ws);
  uint16_t* wqb  = (uint16_t*)(ws + 2 * X_ELEMS);
  uint16_t* wkb  = wqb + W_ELEMS;
  uint16_t* wvb  = wkb + W_ELEMS;
  uint16_t* wob  = wvb + W_ELEMS;
  uint16_t* Qb   = wob + W_ELEMS;
  uint16_t* Kb   = Qb + QKV_ELEMS;
  uint16_t* Vb   = Kb + QKV_ELEMS;
  uint16_t* ctxb = Vb + QKV_ELEMS;

  // 1) fp32 -> bf16 conversions
  {
    const int n1 = (int)X_ELEMS, n2 = (int)W_ELEMS;
    cvt_f32_bf16<<<(n1 + 255) / 256, 256, 0, stream>>>(x, xb, n1);
    cvt_f32_bf16<<<(n2 + 255) / 256, 256, 0, stream>>>(Wq, wqb, n2);
    cvt_f32_bf16<<<(n2 + 255) / 256, 256, 0, stream>>>(Wk, wkb, n2);
    cvt_f32_bf16<<<(n2 + 255) / 256, 256, 0, stream>>>(Wv, wvb, n2);
    cvt_f32_bf16<<<(n2 + 255) / 256, 256, 0, stream>>>(Wo, wob, n2);
  }

  // 2) Q/K/V projections (scatter to [b,h,s,dk] bf16)
  dim3 ggrid(D_MODEL / BN, M_TOT / BM);   // (8, 32)
  gemm_bf16_wmma<<<ggrid, 256, 0, stream>>>(xb, wqb, bq, Qb, nullptr, 0);
  gemm_bf16_wmma<<<ggrid, 256, 0, stream>>>(xb, wkb, bk, Kb, nullptr, 0);
  gemm_bf16_wmma<<<ggrid, 256, 0, stream>>>(xb, wvb, bv, Vb, nullptr, 0);

  // 3) causal flash attention -> ctx bf16 [b, s, d_model]
  attn_flash_wmma<<<dim3(SEQ / 128, BATCH * N_HEAD), 256, 0, stream>>>(
      Qb, Kb, Vb, ctxb);

  // 4) output projection -> fp32 d_out
  gemm_bf16_wmma<<<ggrid, 256, 0, stream>>>(ctxb, wob, bo, nullptr,
                                            (float*)d_out, 1);
}
